// SimpleRetention_35914516529187
// MI455X (gfx1250) — compile-verified, matrix operand order checked by reading, not run on hardware
//
#include <hip/hip_runtime.h>

typedef __bf16 bf16;
typedef __attribute__((ext_vector_type(16))) __bf16 v16bf;
typedef __attribute__((ext_vector_type(8)))  __bf16 v8bf;
typedef __attribute__((ext_vector_type(8)))  float  v8f;
typedef __attribute__((ext_vector_type(4)))  int    v4i;

#define L_SEQ 2048
#define DIM   256
#define NBH   32          // B*H = 8*4
#define TR    64          // rows per block tile
#define XS_STRIDE 264     // 256 + 8 pad (16B-multiple rows)
#define WS_STRIDE 40      // 32 + 8 pad
#define VT_STRIDE 72      // 64 + 8 pad
#define SS_STRIDE 72
#define LOG_GAMMA   (-0.03174869831458030f)   // ln(0.96875)
#define LOG2_10000  (13.287712379549449f)

// ---- fast exp2 (v_exp_f32) ------------------------------------------------
__device__ __forceinline__ float fast_exp2(float x) {
#if __has_builtin(__builtin_amdgcn_exp2f)
  return __builtin_amdgcn_exp2f(x);
#else
  return exp2f(x);
#endif
}

// ---- CDNA5 async Global->LDS path (ASYNCcnt) ------------------------------
#if __has_builtin(__builtin_amdgcn_global_load_async_to_lds_b128) && \
    __has_builtin(__builtin_amdgcn_s_wait_asynccnt)
#define HAVE_ASYNC 1
#define ASYNC_COPY_B128(gsrc, ldst)                                          \
  __builtin_amdgcn_global_load_async_to_lds_b128(                            \
      (v4i*)(gsrc),                                                          \
      (__attribute__((address_space(3))) v4i*)(ldst), 0, 0)
#define ASYNC_WAIT() __builtin_amdgcn_s_wait_asynccnt(0)
#else
#define HAVE_ASYNC 0
#endif

// Load one 16-bit A/B WMMA fragment from LDS (row-major tile).
// Per ISA: lane L holds row (L%16); elements 0-7 at col c0=(L/16)*8, elements
// 8-15 at col c0+16  ->  two ds_load_b128 per fragment.
__device__ __forceinline__ v16bf ld_frag(const bf16* rowbase, int c0) {
  union { v16bf v; v8bf h[2]; } u;
  u.h[0] = *(const v8bf*)(rowbase + c0);
  u.h[1] = *(const v8bf*)(rowbase + c0 + 16);
  return u.v;
}

__device__ __forceinline__ v8f wmma_bf16(v16bf a, v16bf b, v8f c) {
  return __builtin_amdgcn_wmma_f32_16x16x32_bf16(
      /*neg_a=*/false, a, /*neg_b=*/false, b,
      /*c_mod=*/(short)0, c, /*reuse_a=*/false, /*reuse_b=*/false);
}

// ---------------------------------------------------------------------------
// One projection: Y = X_tile @ W, epilogue specialized at compile time.
// MAT: 0 = Q (xPos upscale), 1 = K (xPos downscale), 2 = V (plain).
// ---------------------------------------------------------------------------
template <int MAT>
__device__ __forceinline__ void proj_mat(const float* __restrict__ W,
                                         bf16* __restrict__ dst,
                                         bf16* Xs, bf16* Ws,
                                         int tid, int lrow, int lhalf, int wave,
                                         int r0, size_t base) {
  v8f acc[4][2] = {};

  for (int k0 = 0; k0 < DIM; k0 += 32) {
    // Stage transposed W slice: Ws[n][kk] = W[k0+kk][n]
#pragma unroll
    for (int kk = 0; kk < 32; ++kk)
      Ws[tid * WS_STRIDE + kk] = (bf16)W[(size_t)(k0 + kk) * DIM + tid];
    __syncthreads();

    v16bf a[4];
#pragma unroll
    for (int rt = 0; rt < 4; ++rt)
      a[rt] = ld_frag(Xs + (rt * 16 + lrow) * XS_STRIDE, k0 + lhalf * 8);
#pragma unroll
    for (int jj = 0; jj < 2; ++jj) {
      v16bf b = ld_frag(Ws + ((wave * 2 + jj) * 16 + lrow) * WS_STRIDE, lhalf * 8);
#pragma unroll
      for (int rt = 0; rt < 4; ++rt)
        acc[rt][jj] = wmma_bf16(a[rt], b, acc[rt][jj]);
    }
    __syncthreads();
  }

  // Epilogue. C layout: row n = r0 + rt*16 + e + 8*lhalf, col f = ct*16+lrow.
  if constexpr (MAT == 2) {
#pragma unroll
    for (int jj = 0; jj < 2; ++jj) {
      int f = (wave * 2 + jj) * 16 + lrow;
#pragma unroll
      for (int rt = 0; rt < 4; ++rt)
#pragma unroll
        for (int e = 0; e < 8; ++e) {
          int n = r0 + rt * 16 + e + lhalf * 8;
          dst[base + (size_t)n * DIM + f] = (bf16)acc[rt][jj][e];
        }
    }
  } else {
    // xPos: rotation partner f^1 sits in the adjacent lane.
    constexpr float sgn = (MAT == 1) ? -1.0f : 1.0f;   // K uses downscale
#pragma unroll
    for (int jj = 0; jj < 2; ++jj) {
      int f = (wave * 2 + jj) * 16 + lrow;
      // per-feature constants hoisted out of the row loop
      float sb   = ((float)(f & ~1) + 102.4f) * (1.0f / 358.4f);
      float c1   = __log2f(sb) * (sgn / 512.0f);                   // log2(scale)/512
      float invf = fast_exp2((float)(f >> 1) * (-LOG2_10000 / 128.0f));
#pragma unroll
      for (int rt = 0; rt < 4; ++rt)
#pragma unroll
        for (int e = 0; e < 8; ++e) {
          int n = r0 + rt * 16 + e + lhalf * 8;
          float v = acc[rt][jj][e];
          float p = __shfl_xor(v, 1, 32);
          float sc = fast_exp2((float)n * c1);
          float sv, cv;
          __sincosf((float)n * invf, &sv, &cv);
          float r = (f & 1) ? (v * cv + p * sv) : (v * cv - p * sv);
          dst[base + (size_t)n * DIM + f] = (bf16)(r * sc);
        }
    }
  }
}

// ---------------------------------------------------------------------------
// Kernel 1: Q/K/V projections (bf16 WMMA, f32 accum) with fused xPos epilogue.
// grid = (L/64, B*H), block = 256 (8 waves).
// ---------------------------------------------------------------------------
__global__ __launch_bounds__(256, 2)
void proj_xpos_kernel(const float* __restrict__ X,
                      const float* __restrict__ Wq,
                      const float* __restrict__ Wk,
                      const float* __restrict__ Wv,
                      bf16* __restrict__ Qb,
                      bf16* __restrict__ Kb,
                      bf16* __restrict__ Vb) {
  extern __shared__ __align__(16) char smem[];
  bf16* Xs = (bf16*)smem;                          // 64 x 264
  bf16* Ws = (bf16*)(smem + TR * XS_STRIDE * 2);   // 256 x 40 (W^T k-slice)

  const int tid   = threadIdx.x;
  const int lane  = tid & 31;
  const int wave  = tid >> 5;
  const int lrow  = lane & 15;
  const int lhalf = lane >> 4;
  const int bh = blockIdx.y;
  const int r0 = blockIdx.x * TR;
  const size_t base = (size_t)bh * L_SEQ * DIM;

  // Stage X tile -> bf16 LDS (coalesced float4 reads)
#pragma unroll
  for (int i = 0; i < 16; ++i) {
    int idx = tid + 256 * i;          // 4096 float4 chunks
    int r = idx >> 6, c4 = idx & 63;
    float4 xv = *(const float4*)(X + base + (size_t)(r0 + r) * DIM + c4 * 4);
    bf16* p = Xs + r * XS_STRIDE + c4 * 4;
    p[0] = (bf16)xv.x; p[1] = (bf16)xv.y; p[2] = (bf16)xv.z; p[3] = (bf16)xv.w;
  }
  __syncthreads();

  proj_mat<0>(Wq, Qb, Xs, Ws, tid, lrow, lhalf, wave, r0, base);
  proj_mat<1>(Wk, Kb, Xs, Ws, tid, lrow, lhalf, wave, r0, base);
  proj_mat<2>(Wv, Vb, Xs, Ws, tid, lrow, lhalf, wave, r0, base);
}

// ---------------------------------------------------------------------------
// Kernel 2: retention attention, flash-style over causal 64x64 blocks.
// O accumulators (64 f32 VGPRs/lane) persist across the m loop; S never
// touches HBM. grid = (L/64, B*H), block = 256.
// ---------------------------------------------------------------------------
__global__ __launch_bounds__(256, 2)
void retention_attn_kernel(const bf16* __restrict__ Qb,
                           const bf16* __restrict__ Kb,
                           const bf16* __restrict__ Vb,
                           float* __restrict__ Out) {
  extern __shared__ __align__(16) char smem[];
  bf16* Qs = (bf16*)smem;                                          // 64 x 264
  bf16* Ks = (bf16*)(smem + (size_t)TR * XS_STRIDE * 2);           // 64 x 264
  bf16* Vt = (bf16*)(smem + 2ull * TR * XS_STRIDE * 2);            // 256 x 72 (V^T)
  bf16* Ss = (bf16*)(smem + 2ull * TR * XS_STRIDE * 2
                          + (size_t)DIM * VT_STRIDE * 2);          // 64 x 72

  const int tid   = threadIdx.x;
  const int lane  = tid & 31;
  const int wave  = tid >> 5;
  const int lrow  = lane & 15;
  const int lhalf = lane >> 4;
  const int bh = blockIdx.y;
  const int n0 = blockIdx.x * TR;
  const size_t base = (size_t)bh * L_SEQ * DIM;

  // Stage Q rows once
#pragma unroll
  for (int i = 0; i < 8; ++i) {
    int idx = tid + 256 * i;
    int r = idx >> 5, c8 = idx & 31;
#if HAVE_ASYNC
    ASYNC_COPY_B128(Qb + base + (size_t)(n0 + r) * DIM + c8 * 8,
                    Qs + r * XS_STRIDE + c8 * 8);
#else
    *(v8bf*)(Qs + r * XS_STRIDE + c8 * 8) =
        *(const v8bf*)(Qb + base + (size_t)(n0 + r) * DIM + c8 * 8);
#endif
  }

  v8f o[4][2] = {};
  const int si = wave & 3;        // S row-tile of this wave
  const int sjb = wave >> 2;      // S col-tiles sjb, sjb+2

  for (int m0 = 0; m0 <= n0; m0 += TR) {
    __syncthreads();
    // Stage K (async direct-to-LDS) and V^T (register transpose)
#pragma unroll
    for (int i = 0; i < 8; ++i) {
      int idx = tid + 256 * i;
      int r = idx >> 5, c8 = idx & 31;
#if HAVE_ASYNC
      ASYNC_COPY_B128(Kb + base + (size_t)(m0 + r) * DIM + c8 * 8,
                      Ks + r * XS_STRIDE + c8 * 8);
#else
      *(v8bf*)(Ks + r * XS_STRIDE + c8 * 8) =
          *(const v8bf*)(Kb + base + (size_t)(m0 + r) * DIM + c8 * 8);
#endif
      v8bf vv = *(const v8bf*)(Vb + base + (size_t)(m0 + r) * DIM + c8 * 8);
#pragma unroll
      for (int k = 0; k < 8; ++k)
        Vt[(c8 * 8 + k) * VT_STRIDE + r] = vv[k];
    }
    if (m0 + TR < L_SEQ) {   // prefetch next block -> global_prefetch_b8
      __builtin_prefetch(Kb + base + (size_t)(m0 + TR) * DIM + tid * 8, 0, 1);
      __builtin_prefetch(Vb + base + (size_t)(m0 + TR) * DIM + tid * 8, 0, 1);
    }
#if HAVE_ASYNC
    ASYNC_WAIT();            // s_wait_asynccnt 0 before the barrier
#endif
    __syncthreads();

    // S = Q * K^T : wave computes tiles (si, sjb) and (si, sjb+2)
    v8f sacc[2] = {};
    for (int d0 = 0; d0 < DIM; d0 += 32) {
      v16bf aq = ld_frag(Qs + (si * 16 + lrow) * XS_STRIDE, d0 + lhalf * 8);
#pragma unroll
      for (int u = 0; u < 2; ++u) {
        v16bf bk = ld_frag(Ks + ((sjb + u * 2) * 16 + lrow) * XS_STRIDE, d0 + lhalf * 8);
        sacc[u] = wmma_bf16(aq, bk, sacc[u]);
      }
    }

    // Apply causal decay gamma^(n-m) and spill S (bf16) to LDS
#pragma unroll
    for (int u = 0; u < 2; ++u) {
      int sj = sjb + u * 2;
      int m = m0 + sj * 16 + lrow;
#pragma unroll
      for (int e = 0; e < 8; ++e) {
        int mr = si * 16 + e + lhalf * 8;
        int n = n0 + mr;
        float d = (n >= m) ? __expf((float)(n - m) * LOG_GAMMA) : 0.0f;
        Ss[mr * SS_STRIDE + sj * 16 + lrow] = (bf16)(sacc[u][e] * d);
      }
    }
    __syncthreads();

    // O += S * V  (K-dim = 64, two WMMA steps)
    for (int k0 = 0; k0 < TR; k0 += 32) {
      v16bf as[4];
#pragma unroll
      for (int rt = 0; rt < 4; ++rt)
        as[rt] = ld_frag(Ss + (rt * 16 + lrow) * SS_STRIDE, k0 + lhalf * 8);
#pragma unroll
      for (int jj = 0; jj < 2; ++jj) {
        v16bf bv = ld_frag(Vt + ((wave * 2 + jj) * 16 + lrow) * VT_STRIDE, k0 + lhalf * 8);
#pragma unroll
        for (int rt = 0; rt < 4; ++rt)
          o[rt][jj] = wmma_bf16(as[rt], bv, o[rt][jj]);
      }
    }
  }

  // Write O (f32)
#pragma unroll
  for (int rt = 0; rt < 4; ++rt) {
#pragma unroll
    for (int jj = 0; jj < 2; ++jj) {
      int f = (wave * 2 + jj) * 16 + lrow;
#pragma unroll
      for (int e = 0; e < 8; ++e) {
        int n = n0 + rt * 16 + e + lhalf * 8;
        Out[base + (size_t)n * DIM + f] = o[rt][jj][e];
      }
    }
  }
}

// ---------------------------------------------------------------------------
static constexpr size_t QKV_BYTES = (size_t)NBH * L_SEQ * DIM * 2; // 32 MiB each

extern "C" void kernel_launch(void* const* d_in, const int* in_sizes, int n_in,
                              void* d_out, int out_size, void* d_ws, size_t ws_size,
                              hipStream_t stream) {
  const float* X  = (const float*)d_in[0];
  const float* Wq = (const float*)d_in[1];
  const float* Wk = (const float*)d_in[2];
  const float* Wv = (const float*)d_in[3];
  float* Out = (float*)d_out;

  char* ws = (char*)d_ws;
  bf16* Qb = (bf16*)ws;
  bf16* Kb = (bf16*)(ws + QKV_BYTES);
  bf16* Vb = (bf16*)(ws + 2 * QKV_BYTES);

  dim3 grid(L_SEQ / TR, NBH);
  size_t lds_p = (size_t)TR * XS_STRIDE * 2 + (size_t)DIM * WS_STRIDE * 2;
  size_t lds_a = 2ull * TR * XS_STRIDE * 2 + (size_t)DIM * VT_STRIDE * 2
               + (size_t)TR * SS_STRIDE * 2;

  proj_xpos_kernel<<<grid, dim3(256), lds_p, stream>>>(X, Wq, Wk, Wv, Qb, Kb, Vb);
  retention_attn_kernel<<<grid, dim3(256), lds_a, stream>>>(Qb, Kb, Vb, Out);
}
